// LocalSelfAttention_8426725835120
// MI455X (gfx1250) — matrix operand order repaired and checked by reading, not compiled, not run on hardware
//
#include <hip/hip_runtime.h>
#include <hip/hip_bf16.h>
#include <math.h>

// ---------------------------------------------------------------------------
// LocalSelfAttention for MI455X (gfx1250, wave32, WMMA)
//   B=128, H=4, SEQ=256, E=64, D=64, LH=64  ->  per-head GEMMs M=128,N=4096,K=4096
// ---------------------------------------------------------------------------

#define BATCH   128
#define HEADS   4
#define NDIM    4096        // LH*D
#define KDIM    4096        // LH*E
#define LH      64
#define DD      64
#define SEQLEN  256
#define LDSROW  40          // padded row stride (halfwords) -> conflict-free frags

typedef __attribute__((ext_vector_type(16))) __bf16 v16bf;
typedef __attribute__((ext_vector_type(8)))  float  v8f;

union FragU { uint4 q[2]; v16bf v; };

#if defined(__HIP_DEVICE_COMPILE__) && __has_builtin(__builtin_amdgcn_global_load_async_to_lds_b128)
#define USE_ASYNC_LDS 1
// exact pointee type expected by the builtin: int __vector(4)
typedef int v4i_vs __attribute__((vector_size(16)));
typedef __attribute__((address_space(1))) v4i_vs gas_v4i;   // global
typedef __attribute__((address_space(3))) v4i_vs las_v4i;   // LDS
#else
#define USE_ASYNC_LDS 0
#endif

// f32 pair -> packed bf16 (RNE). Prefer the native v_cvt_pk_bf16_f32.
#if defined(__HIP_DEVICE_COMPILE__) && __has_builtin(__builtin_amdgcn_cvt_pk_bf16_f32)
typedef __attribute__((ext_vector_type(2))) __bf16 v2bf;
__device__ __forceinline__ unsigned pk_bf16(float a, float b) {
    union { v2bf v; unsigned u; } cv;
    cv.v = __builtin_amdgcn_cvt_pk_bf16_f32(a, b);
    return cv.u;
}
#else
__device__ __forceinline__ unsigned pk_bf16(float a, float b) {
    unsigned ua = __float_as_uint(a);
    unsigned ub = __float_as_uint(b);
    ua = (ua + 0x7FFFu + ((ua >> 16) & 1u)) >> 16;
    ub = (ub + 0x7FFFu + ((ub >> 16) & 1u)) >> 16;
    return (ua & 0xFFFFu) | (ub << 16);
}
#endif

// ---------------------------------------------------------------------------
// Kernel 1: per-head projection GEMM  C[b,o] = sum_i X[b,i]*W[o,i] + bias[o]
//   grid: (NDIM/128, HEADS, 3 projections), block: 256 threads (8 waves)
//   Double-buffered LDS (bf16 tiles), one barrier per 32-deep K-step.
// ---------------------------------------------------------------------------
__global__ __launch_bounds__(256)
void lsa_proj_gemm(const float* __restrict__ seq,
                   const float* __restrict__ WQ, const float* __restrict__ bQ,
                   const float* __restrict__ WK, const float* __restrict__ bK,
                   const float* __restrict__ WV, const float* __restrict__ bV,
                   float* __restrict__ qkv)     // [3][H][B][NDIM]
{
    __shared__ __align__(16) unsigned short As[2][128 * LDSROW];  // X tiles, bf16
    __shared__ __align__(16) unsigned short Bs[2][128 * LDSROW];  // W tiles, bf16

    const int tid  = threadIdx.x;
    const int lane = tid & 31;
    const int wid  = tid >> 5;

    const int h     = blockIdx.y;
    const int p     = blockIdx.z;                 // 0=Q 1=K 2=V
    const int ntile = blockIdx.x * 128;

    const float* W    = (p == 0) ? WQ : ((p == 1) ? WK : WV);
    const float* bias = (p == 0) ? bQ : ((p == 1) ? bK : bV);

    // staging assignment: 2 threads per row, 16 f32 each
    const int srow = tid >> 1;
    const int scol = (tid & 1) * 16;
    const float* xg = seq + (size_t)srow * (SEQLEN * 64) + (size_t)h * KDIM + scol;
    const float* wg = W + (size_t)h * NDIM * KDIM + (size_t)(ntile + srow) * KDIM + scol;

    // wave tiling: 2 waves over M (64 rows each), 4 waves over N (32 cols each)
    const int waveM   = wid & 1;
    const int waveN   = wid >> 1;
    const int mBase   = waveM * 64;
    const int nBase   = waveN * 32;
    const int halfSel = lane >> 4;     // 0: lanes 0-15, 1: lanes 16-31
    const int l16     = lane & 15;

    v8f acc[4][2];
#pragma unroll
    for (int mt = 0; mt < 4; ++mt)
#pragma unroll
        for (int nt = 0; nt < 2; ++nt)
            acc[mt][nt] = (v8f){0.f, 0.f, 0.f, 0.f, 0.f, 0.f, 0.f, 0.f};

    // convert 16 f32 -> 16 bf16 and store as 2 x b128 into buffer `buf`
    auto stageStore = [&](int buf, const float4 xv[4], const float4 wv[4]) {
        unsigned short* asp = &As[buf][srow * LDSROW + scol];
        unsigned short* bsp = &Bs[buf][srow * LDSROW + scol];
        uint4 lo, hi;
        lo = make_uint4(pk_bf16(xv[0].x, xv[0].y), pk_bf16(xv[0].z, xv[0].w),
                        pk_bf16(xv[1].x, xv[1].y), pk_bf16(xv[1].z, xv[1].w));
        hi = make_uint4(pk_bf16(xv[2].x, xv[2].y), pk_bf16(xv[2].z, xv[2].w),
                        pk_bf16(xv[3].x, xv[3].y), pk_bf16(xv[3].z, xv[3].w));
        *(uint4*)asp       = lo;
        *(uint4*)(asp + 8) = hi;
        lo = make_uint4(pk_bf16(wv[0].x, wv[0].y), pk_bf16(wv[0].z, wv[0].w),
                        pk_bf16(wv[1].x, wv[1].y), pk_bf16(wv[1].z, wv[1].w));
        hi = make_uint4(pk_bf16(wv[2].x, wv[2].y), pk_bf16(wv[2].z, wv[2].w),
                        pk_bf16(wv[3].x, wv[3].y), pk_bf16(wv[3].z, wv[3].w));
        *(uint4*)bsp       = lo;
        *(uint4*)(bsp + 8) = hi;
    };

    // ---- prologue: stage K-slab 0 into buffer 0 ---------------------------
    float4 xv[4], wv[4];
#pragma unroll
    for (int i = 0; i < 4; ++i) {
        xv[i] = *(const float4*)(xg + i * 4);
        wv[i] = *(const float4*)(wg + i * 4);
    }
    stageStore(0, xv, wv);

    for (int k0 = 0; k0 < KDIM; k0 += 32) {
        const int  cur  = (k0 >> 5) & 1;
        const bool more = (k0 + 32) < KDIM;

        if (more) {      // fetch next K-slab (overlaps this step's WMMA)
#pragma unroll
            for (int i = 0; i < 4; ++i) {
                xv[i] = *(const float4*)(xg + k0 + 32 + i * 4);
                wv[i] = *(const float4*)(wg + k0 + 32 + i * 4);
            }
            __builtin_prefetch(wg + k0 + 96, 0, 1);   // global_prefetch_b8
            __builtin_prefetch(xg + k0 + 96, 0, 1);
        }

        __syncthreads();   // tile `cur` visible; tile `cur^1` free to overwrite

        // ---- B fragments (32x16 bf16): lane<16 -> n=l16,k {0..7,16..23}; else k {8..15,24..31}
        v16bf bf[2];
#pragma unroll
        for (int nt = 0; nt < 2; ++nt) {
            const uint4* pB =
                (const uint4*)&Bs[cur][(nBase + nt * 16 + l16) * LDSROW + halfSel * 8];
            FragU u; u.q[0] = pB[0]; u.q[1] = pB[2];
            bf[nt] = u.v;
        }
        // ---- A fragments (16x32 bf16): lane<16 -> m=l16, K {0..7,16..23};
        //                                lane>=16 -> m=l16, K {8..15,24..31}
#pragma unroll
        for (int mt = 0; mt < 4; ++mt) {
            const unsigned short* pa =
                &As[cur][(mBase + mt * 16 + l16) * LDSROW + halfSel * 8];
            FragU u;
            u.q[0] = *(const uint4*)pa;
            u.q[1] = *(const uint4*)(pa + 16);
            const v16bf af = u.v;
#pragma unroll
            for (int nt = 0; nt < 2; ++nt)
                acc[mt][nt] = __builtin_amdgcn_wmma_f32_16x16x32_bf16(
                    false, af, false, bf[nt], (short)0, acc[mt][nt], false, false);
        }

        if (more) stageStore(cur ^ 1, xv, wv);
    }

    // ---- epilogue: C layout VGPR r: lanes0-15 M=r, lanes16-31 M=8+r ------
    float* outBase = qkv + (size_t)(p * HEADS + h) * BATCH * NDIM;
#pragma unroll
    for (int mt = 0; mt < 4; ++mt) {
#pragma unroll
        for (int nt = 0; nt < 2; ++nt) {
            const int n  = ntile + nBase + nt * 16 + l16;
            const float bb = bias[(size_t)h * NDIM + n];
            float* cptr = outBase +
                (size_t)(mBase + mt * 16 + halfSel * 8) * NDIM + n;
#pragma unroll
            for (int r = 0; r < 8; ++r)
                cptr[(size_t)r * NDIM] = acc[mt][nt][r] + bb;
        }
    }
}

// ---------------------------------------------------------------------------
// Kernel 2: per-(b,h) local attention on 64x64 tiles.
//   scores = Q K^T + causal mask; softmax over the QUERY axis (axis=2,
//   faithful to reference); Z = attn V * (1/sqrt(D)).
//   Q/K/V tiles are staged with CDNA5 async global->LDS copies when available.
// ---------------------------------------------------------------------------
__global__ __launch_bounds__(256)
void lsa_attention(const float* __restrict__ qkv, float* __restrict__ out)
{
    __shared__ __align__(16) float Qs[LH * DD];   // reused for V after scores
    __shared__ __align__(16) float Ks[LH * DD];
    __shared__ float Ss[LH * LH];

    const int tid = threadIdx.x;
    const int b   = blockIdx.x >> 2;
    const int h   = blockIdx.x & 3;

    const size_t plane = (size_t)HEADS * BATCH * NDIM;
    const float* Qg = qkv +             ((size_t)h * BATCH + b) * NDIM;
    const float* Kg = qkv + plane     + ((size_t)h * BATCH + b) * NDIM;
    const float* Vg = qkv + 2 * plane + ((size_t)h * BATCH + b) * NDIM;

#if USE_ASYNC_LDS
    {   // 16 KB per tile; 256 threads x 16 B x 4 sweeps, ASYNCcnt-tracked
        const char* gq = (const char*)Qg + tid * 16;
        const char* gk = (const char*)Kg + tid * 16;
        char* lq = (char*)Qs + tid * 16;
        char* lk = (char*)Ks + tid * 16;
#pragma unroll
        for (int i = 0; i < 4; ++i) {
            __builtin_amdgcn_global_load_async_to_lds_b128(
                (gas_v4i*)(gq + i * 4096), (las_v4i*)(lq + i * 4096), 0, 0);
            __builtin_amdgcn_global_load_async_to_lds_b128(
                (gas_v4i*)(gk + i * 4096), (las_v4i*)(lk + i * 4096), 0, 0);
        }
#if __has_builtin(__builtin_amdgcn_s_wait_asynccnt)
        __builtin_amdgcn_s_wait_asynccnt(0);
#else
        asm volatile("s_wait_asynccnt 0x0" ::: "memory");
#endif
    }
#else
    for (int i = tid; i < LH * DD; i += 256) { Qs[i] = Qg[i]; Ks[i] = Kg[i]; }
#endif
    __syncthreads();

    // scores: 4 threads per query row, 16 keys each
    {
        const int q  = tid >> 2;
        const int kb = (tid & 3) * 16;
#pragma unroll 4
        for (int kk = 0; kk < 16; ++kk) {
            const int k = kb + kk;
            float s = 0.f;
#pragma unroll 8
            for (int d = 0; d < DD; ++d) s += Qs[q * DD + d] * Ks[k * DD + d];
            if (k > q) s += -1e20f;          // triu(..., diagonal=1) mask
            Ss[q * LH + k] = s;
        }
    }
    __syncthreads();

    // load V into Qs (Q no longer needed); softmax over q by threads 0..63
#if USE_ASYNC_LDS
    {
        const char* gv = (const char*)Vg + tid * 16;
        char* lv = (char*)Qs + tid * 16;
#pragma unroll
        for (int i = 0; i < 4; ++i)
            __builtin_amdgcn_global_load_async_to_lds_b128(
                (gas_v4i*)(gv + i * 4096), (las_v4i*)(lv + i * 4096), 0, 0);
    }
#else
    for (int i = tid; i < LH * DD; i += 256) Qs[i] = Vg[i];
#endif
    if (tid < LH) {
        const int k = tid;
        float mx = -3.0e38f;
        for (int q = 0; q < LH; ++q) mx = fmaxf(mx, Ss[q * LH + k]);
        float sum = 0.f;
        for (int q = 0; q < LH; ++q) {
            const float e = __expf(Ss[q * LH + k] - mx);
            Ss[q * LH + k] = e;
            sum += e;
        }
        const float inv = 1.f / sum;
        for (int q = 0; q < LH; ++q) Ss[q * LH + k] *= inv;
    }
#if USE_ASYNC_LDS
#if __has_builtin(__builtin_amdgcn_s_wait_asynccnt)
    __builtin_amdgcn_s_wait_asynccnt(0);
#else
    asm volatile("s_wait_asynccnt 0x0" ::: "memory");
#endif
#endif
    __syncthreads();

    // Z = attn * V * (1/8); 4 threads per query row, 16 d each
    {
        const int q  = tid >> 2;
        const int db = (tid & 3) * 16;
#pragma unroll 4
        for (int j = 0; j < 16; ++j) {
            const int d = db + j;
            float z = 0.f;
#pragma unroll 8
            for (int k = 0; k < LH; ++k) z += Ss[q * LH + k] * Qs[k * DD + d];
            out[((size_t)b * SEQLEN + h * LH + q) * DD + d] = z * 0.125f;
        }
    }
}

// ---------------------------------------------------------------------------
extern "C" void kernel_launch(void* const* d_in, const int* in_sizes, int n_in,
                              void* d_out, int out_size, void* d_ws, size_t ws_size,
                              hipStream_t stream)
{
    const float* seq = (const float*)d_in[0];
    const float* WQ  = (const float*)d_in[1];
    const float* bQ  = (const float*)d_in[2];
    const float* WK  = (const float*)d_in[3];
    const float* bK  = (const float*)d_in[4];
    const float* WV  = (const float*)d_in[5];
    const float* bV  = (const float*)d_in[6];

    float* qkv = (float*)d_ws;                  // 3*H*B*NDIM f32 = 25.2 MB
    float* out = (float*)d_out;

    dim3 gridGemm(NDIM / 128, HEADS, 3);
    lsa_proj_gemm<<<gridGemm, 256, 0, stream>>>(seq, WQ, bQ, WK, bK, WV, bV, qkv);

    lsa_attention<<<BATCH * HEADS, 256, 0, stream>>>(qkv, out);
}
